// kernel_generated_1_62904091017529
// MI455X (gfx1250) — matrix operand order repaired and checked by reading, not compiled, not run on hardware
//
#include <hip/hip_runtime.h>

// ---------------------------------------------------------------------------
// y[b,i,n,m'] = C_m[(n-1)%28, i]*w0[i,1] + C_m[n, i]*w0[i,0],  m' = (m+1)%28
// C_m[n,i]   = sum_{j,k} xp[b, j*28+m, n+k] * w1[j,k,i]   (K = 192 contraction)
// Per (b,m): 32x192 (padded) x 192x128 GEMM in bf16 WMMA, f32 accumulate.
// x-slices (64 rows x 28 f32, row stride 784 f32) are staged global->LDS by
// the Tensor Data Mover (TENSOR_LOAD_TO_LDS, TENSORcnt-tracked), double
// buffered, so the im2col build never touches global memory.
// ---------------------------------------------------------------------------

typedef __attribute__((ext_vector_type(16))) __bf16 v16bf;
typedef __attribute__((ext_vector_type(8)))  float  v8f;
typedef __attribute__((ext_vector_type(4)))  float  f32x4;
typedef __attribute__((ext_vector_type(4)))  int    v4i;
typedef __attribute__((ext_vector_type(4)))  unsigned int u32x4;
typedef __attribute__((ext_vector_type(8)))  int    i32x8;
typedef __attribute__((ext_vector_type(4)))  int    i32x4;

#define A_STRIDE 200   // ushorts per A row (400 B): 16B-aligned, conflict-free

#if __has_builtin(__builtin_amdgcn_tensor_load_to_lds)
#define HAVE_TDM 1
#endif

static __device__ __forceinline__ unsigned short f32_to_bf16_rne(float f) {
  unsigned u = __builtin_bit_cast(unsigned, f);
  u += 0x7FFFu + ((u >> 16) & 1u);
  return (unsigned short)(u >> 16);
}

// ---- fallback: per-lane async global->LDS copy (ASYNCcnt-tracked) ---------
static __device__ __forceinline__ void async_copy_b128(const void* g, void* l) {
#if __has_builtin(__builtin_amdgcn_global_load_async_to_lds_b128)
  __builtin_amdgcn_global_load_async_to_lds_b128(
      (__attribute__((address_space(1))) v4i*)g,
      (__attribute__((address_space(3))) v4i*)l, 0, 0);
#else
  unsigned la = (unsigned)(unsigned long long)l;
  asm volatile("global_load_async_to_lds_b128 %0, %1, off"
               :: "v"(la), "v"(g) : "memory");
#endif
}

static __device__ __forceinline__ void wait_async0() {
#if __has_builtin(__builtin_amdgcn_s_wait_asynccnt)
  __builtin_amdgcn_s_wait_asynccnt(0);
#else
  asm volatile("s_wait_asynccnt 0x0" ::: "memory");
#endif
}

static __device__ __forceinline__ void wait_tensor0() {
#if __has_builtin(__builtin_amdgcn_s_wait_tensorcnt)
  __builtin_amdgcn_s_wait_tensorcnt(0);
#else
  asm volatile("s_wait_tensorcnt 0x0" ::: "memory");
#endif
}

#ifdef HAVE_TDM
// One TDM descriptor: 2D tile, 64 lines x 28 f32, line stride 784 f32,
// packed contiguously into LDS at lds_dst.  (CDNA5 ISA)
static __device__ __forceinline__ void tdm_stage_slice(const float* gtile,
                                                       void* lds_dst) {
  unsigned lds_addr = (unsigned)(unsigned long long)lds_dst;  // AS3 offset
  unsigned long long ga = (unsigned long long)gtile;
  u32x4 g0;
  g0[0] = 1u;                                        // count=1, user mode
  g0[1] = lds_addr;                                  // lds_addr [63:32]
  g0[2] = (unsigned)(ga & 0xFFFFFFFFu);              // global_addr low
  g0[3] = (unsigned)((ga >> 32) & 0x1FFFFFFu)        // global_addr [56:32]
        | (2u << 30);                                // type=2 ("image")
  i32x8 g1;
  g1[0] = (int)(2u << 16);    // wg_mask=0, data_size=2 (4B), no pad/iterate
  g1[1] = (int)(28u << 16);   // tensor_dim0 = 28      (bits 79:48 -> low16)
  g1[2] = (int)(64u << 16);   // tensor_dim1 = 64      (bits 111:80 -> low16)
  g1[3] = (int)(28u << 16);   // tile_dim0   = 28      (bits 127:112)
  g1[4] = 64;                 // tile_dim1   = 64, tile_dim2 = 0
  g1[5] = 784;                // tensor_dim0_stride = 784 (bits 207:160)
  g1[6] = 0;                  // stride hi / tensor_dim1_stride lo
  g1[7] = 0;
  i32x4 z4 = {0, 0, 0, 0};                 // groups 2/3 unused (<=2D tensor)
  i32x8 z8 = {0, 0, 0, 0, 0, 0, 0, 0};     // trailing group (6-arg variant)
  __builtin_amdgcn_tensor_load_to_lds(g0, g1, z4, z4, z8, 0);
}
#endif

// stage slice m of batch-slab xb into LDS dst (1792 f32, contiguous)
static __device__ __forceinline__ void stage_slice(const float* xb, int m,
                                                   float* dst, int tid) {
#ifdef HAVE_TDM
  if (tid < 32) tdm_stage_slice(xb + m * 28, dst);   // one wave issues the DMA
#else
  for (int c = tid; c < 448; c += 256) {
    int j = c / 7, q = c - 7 * j;
    async_copy_b128(xb + (j * 28 + m) * 28 + q * 4, dst + j * 28 + q * 4);
  }
#endif
}

static __device__ __forceinline__ void stage_wait(int tid) {
#ifdef HAVE_TDM
  if (tid < 32) wait_tensor0();    // issuing wave drains its TENSORcnt
#else
  wait_async0();
#endif
}

// w1 (64,3,128) f32  ->  Bt[i=128][kl=192] bf16   (Bt[i*192 + 3j+k] = w1[j,k,i])
__global__ void prep_w1_bf16(const float* __restrict__ w1,
                             unsigned short* __restrict__ bt) {
  int t = blockIdx.x * blockDim.x + threadIdx.x;
  for (int e = t; e < 128 * 192; e += gridDim.x * blockDim.x) {
    int i  = e / 192;
    int kl = e - i * 192;
    bt[e] = f32_to_bf16_rne(w1[kl * 128 + i]);
  }
}

union Frag { v16bf v; f32x4 q[2]; };

__launch_bounds__(256)
__global__ void fused_conv_gemm(const float* __restrict__ x,
                                const float* __restrict__ w0,
                                const unsigned short* __restrict__ bt,
                                float* __restrict__ y) {
  __shared__ __align__(16) unsigned short sA[32 * A_STRIDE]; // 12.8 KB
  __shared__ __align__(16) float xs[2][1792];                // 14.3 KB staged x
  __shared__ float sC[8][512];                               // 16 KB, per-wave

  const int b    = blockIdx.x;
  const int tid  = threadIdx.x;
  const int lane = tid & 31;
  const int wave = tid >> 5;     // 8 waves: one 16-wide i-block each
  const int hi   = lane >> 4;    // lane half (K-half selector in WMMA layouts)
  const int il   = lane & 15;
  const int i    = wave * 16 + il;   // output feature column for this lane

  const float w00 = w0[2 * i + 0];
  const float w01 = w0[2 * i + 1];

  const float* xb = x + (size_t)b * (1792 * 28);
  float*       yb = y + (size_t)b * (128 * 28 * 28);

  // ---- B fragments (32x16 bf16 each), loaded once, live in VGPRs ----------
  Frag Bf[6];
#pragma unroll
  for (int kb = 0; kb < 6; ++kb) {
    const unsigned short* bp = bt + i * 192 + kb * 32 + hi * 16;
    Bf[kb].q[0] = *(const f32x4*)(bp);
    Bf[kb].q[1] = *(const f32x4*)(bp + 8);
  }

  // ---- stage slice m=0 ----------------------------------------------------
  stage_slice(xb, 0, xs[0], tid);
  stage_wait(tid);
  __syncthreads();

  // build-phase work assignment: fixed (n, j-block) per thread
  const int bn = tid >> 3;         // n row 0..31 (28..31 -> zero pad)
  const int jb = tid & 7;          // j block: j = jb*8 .. jb*8+7

  for (int m = 0; m < 28; ++m) {
    const int cur = m & 1;

    // ---- kick off stage of slice m+1 (no wait yet) ------------------------
    if (m < 27) stage_slice(xb, m + 1, xs[1 - cur], tid);

    // ---- build A_m (im2col, bf16) in LDS from staged slice ----------------
    {
      const float* xc = xs[cur];
      union { unsigned short s[24]; f32x4 q[3]; } pk;
#pragma unroll
      for (int jj = 0; jj < 8; ++jj) {
        int j = jb * 8 + jj;
#pragma unroll
        for (int k = 0; k < 3; ++k) {
          int tt = bn + k - 1;
          float v = (bn < 28 && (unsigned)tt < 28u) ? xc[j * 28 + tt] : 0.0f;
          pk.s[jj * 3 + k] = f32_to_bf16_rne(v);
        }
      }
      f32x4* dst = (f32x4*)((unsigned char*)sA + bn * 400 + jb * 48);
      dst[0] = pk.q[0];
      dst[1] = pk.q[1];
      dst[2] = pk.q[2];
    }
    __syncthreads();   // A_m visible to all waves

    // ---- GEMM: 2 M-tiles x 1 N-tile per wave, K = 6 x 32 ------------------
    Frag A0[6], A1[6];
#pragma unroll
    for (int kb = 0; kb < 6; ++kb) {
      const unsigned short* ap0 = sA + il * A_STRIDE + kb * 32 + hi * 16;
      A0[kb].q[0] = *(const f32x4*)(ap0);
      A0[kb].q[1] = *(const f32x4*)(ap0 + 8);
      const unsigned short* ap1 = ap0 + 16 * A_STRIDE;
      A1[kb].q[0] = *(const f32x4*)(ap1);
      A1[kb].q[1] = *(const f32x4*)(ap1 + 8);
    }
    v8f acc0 = {};  // n rows  0..15
    v8f acc1 = {};  // n rows 16..31 (28..31 zero pad)
#pragma unroll
    for (int kb = 0; kb < 6; ++kb) {
      acc0 = __builtin_amdgcn_wmma_f32_16x16x32_bf16(
                 false, A0[kb].v, false, Bf[kb].v, (short)0, acc0, false, false);
      acc1 = __builtin_amdgcn_wmma_f32_16x16x32_bf16(
                 false, A1[kb].v, false, Bf[kb].v, (short)0, acc1, false, false);
    }

    // ---- spill C tile to per-wave LDS (same-wave DS ops are in-order) -----
    float* cw = sC[wave];
#pragma unroll
    for (int r = 0; r < 8; ++r) {
      cw[(r + 8 * hi) * 16 + il]      = acc0[r];
      cw[(16 + r + 8 * hi) * 16 + il] = acc1[r];
    }

    // ---- epilogue: two-tap combine over wrapped n, store y ----------------
    const int mp = (m + 1) % 28;   // roll along m-axis
#pragma unroll
    for (int itn = 0; itn < 14; ++itn) {
      int n    = 2 * itn + hi;                      // 0..27, each once
      float c0 = cw[n * 16 + il];
      float c1 = cw[((n + 27) % 28) * 16 + il];     // (n-1) mod 28
      yb[((size_t)i * 28 + n) * 28 + mp] = c1 * w01 + c0 * w00;
    }

    // ---- close the pipeline stage ----------------------------------------
    stage_wait(tid);   // slice m+1 fully in LDS (issuing wave drains counter)
    __syncthreads();   // ...and visible to every wave; sA free to overwrite
  }
}

extern "C" void kernel_launch(void* const* d_in, const int* in_sizes, int n_in,
                              void* d_out, int out_size, void* d_ws, size_t ws_size,
                              hipStream_t stream) {
  const float* x  = (const float*)d_in[0];   // (1024, 1792, 28) f32
  const float* w0 = (const float*)d_in[1];   // (128, 2) f32
  const float* w1 = (const float*)d_in[2];   // (64, 3, 128) f32
  float* y = (float*)d_out;                  // (1024, 128, 28, 28) f32
  unsigned short* bt = (unsigned short*)d_ws;  // 128*192 bf16 = 48 KB

  prep_w1_bf16<<<96, 256, 0, stream>>>(w1, bt);
  fused_conv_gemm<<<1024, 256, 0, stream>>>(x, w0, bt, y);
}
